// SoftAttention_21784074126155
// MI455X (gfx1250) — matrix operand order repaired and checked by reading, not compiled, not run on hardware
//
#include <hip/hip_runtime.h>
#include <hip/hip_bf16.h>

typedef __attribute__((ext_vector_type(16))) __bf16 v16bf;
typedef __attribute__((ext_vector_type(8)))  __bf16 v8bf;
typedef __attribute__((ext_vector_type(8)))  float  v8f;
typedef __attribute__((ext_vector_type(4)))  float  v4f;

static constexpr int NSTEPS = 256;
static constexpr int BATCH  = 1024;
static constexpr int DIM    = 256;
static constexpr int BT     = 16;              // batch tile per block
static constexpr int NSPLIT = 8;               // N-chunks (parallelism over steps)
static constexpr int NCHUNK = NSTEPS / NSPLIT; // 32
static constexpr int GBS    = 264;             // padded LDS row stride (bf16 elems)

// workspace layout in float elements
static constexpr size_t WS_Q     = 0;
static constexpr size_t WS_M     = WS_Q + (size_t)BATCH * DIM;          // logits [B][N]
static constexpr size_t WS_Z     = WS_M + (size_t)BATCH * NSTEPS;       // z partials [NSPLIT][B][D]
static constexpr size_t WS_STATS = WS_Z + (size_t)NSPLIT * BATCH * DIM; // (max,sum) [NSPLIT][B][2]
static constexpr size_t WS_WKBF  = WS_STATS + (size_t)NSPLIT * BATCH * 2; // wk bf16 (D*D halfs)

// ---------------------------------------------------------------- wk -> bf16
__global__ void k_conv_wk(const float* __restrict__ wk, __bf16* __restrict__ out) {
    int i = blockIdx.x * 256 + threadIdx.x;
    out[i] = (__bf16)wk[i];
}

// ---------------------------------------------------------------- q = h_t @ wq^T + b (fp32 exact)
__global__ __launch_bounds__(256)
void k_q(const float* __restrict__ h, const float* __restrict__ wq,
         const float* __restrict__ wqb, float* __restrict__ q) {
    __shared__ float hs[BT * DIM];
    const int b0 = blockIdx.x * BT;
    for (int i = threadIdx.x; i < BT * DIM; i += 256)
        hs[i] = h[(size_t)b0 * DIM + i];
    __syncthreads();
    const int d = threadIdx.x;
    const float* wr = wq + (size_t)d * DIM;
    float acc[BT];
#pragma unroll
    for (int b = 0; b < BT; ++b) acc[b] = 0.f;
    for (int e = 0; e < DIM; e += 4) {
        v4f wv = *(const v4f*)(wr + e);
#pragma unroll
        for (int b = 0; b < BT; ++b) {
            v4f hv = *(const v4f*)(&hs[b * DIM + e]);
            acc[b] += hv.x * wv.x + hv.y * wv.y + hv.z * wv.z + hv.w * wv.w;
        }
    }
    const float bias = wqb[d];
#pragma unroll
    for (int b = 0; b < BT; ++b)
        q[(size_t)(b0 + b) * DIM + d] = acc[b] + bias;
}

// ---------------------------------------------------------------- fused main kernel
// One pass over G. Per n: WMMA (bf16) pre-activation, tanh, wg-dot, online
// softmax stats, and online-rescaled z accumulation from fp32 G registers.
__global__ __launch_bounds__(512)
void k_main(const float* __restrict__ G, const float* __restrict__ q,
            const float* __restrict__ wkb, const __bf16* __restrict__ wkbf,
            const float* __restrict__ wg,
            float* __restrict__ ws_m, float* __restrict__ ws_z,
            float* __restrict__ ws_stats) {
    __shared__ __bf16 Gb[BT * GBS];      // bf16 A tile (padded)
    __shared__ float  qs[BT * DIM];      // q + wk_b
    __shared__ float  zsh[BT * DIM];     // z accumulator
    __shared__ float  mpart[16][BT];     // per-wave logit partials (deterministic reduce)
    __shared__ float  runmax[BT], runsum[BT], wsh[BT], rfsh[BT];

    const int tid   = threadIdx.x;
    const int w     = tid >> 5;          // wave index 0..15, owns 16 output cols + 1 batch row of z
    const int lane  = tid & 31;
    const int l15   = lane & 15;
    const int hi    = lane >> 4;
    const int b0    = (blockIdx.x & 63) * BT;
    const int chunk = blockIdx.x >> 6;
    const int n0    = chunk * NCHUNK;
    const int ncol  = w * 16 + l15;      // this lane's output column (e)

    for (int i = tid; i < BT * DIM; i += 512) {
        int e = i & 255;
        qs[i]  = q[(size_t)(b0 + (i >> 8)) * DIM + e] + wkb[e];
        zsh[i] = 0.f;
    }
    if (tid < BT) { runmax[tid] = -__builtin_inff(); runsum[tid] = 0.f; }

    // preload wk B-fragments for this wave's 16 columns: B[k][n] = wk[n][k]
    union V16 { v16bf v; v8bf h[2]; };
    V16 bfrag[8];
#pragma unroll
    for (int j = 0; j < 8; ++j) {
        const __bf16* p = wkbf + (size_t)ncol * DIM + 32 * j + hi * 8;
        bfrag[j].h[0] = *(const v8bf*)(p);
        bfrag[j].h[1] = *(const v8bf*)(p + 16);
    }
    const float wgv = wg[ncol];

    // first G tile: wave w holds batch row (b0+w), lanes cover d = lane*8..+7
    const int gd0 = lane * 8;
    {
        const float* gp = G + ((size_t)n0 * BATCH + (b0 + w)) * DIM + gd0;
        // loaded below into g0/g1
    }
    const float* gp0 = G + ((size_t)n0 * BATCH + (b0 + w)) * DIM + gd0;
    v4f g0 = *(const v4f*)(gp0);
    v4f g1 = *(const v4f*)(gp0 + 4);

    __syncthreads();

    for (int ni = 0; ni < NCHUNK; ++ni) {
        const int n = n0 + ni;

        // stage bf16 A tile into LDS
        __bf16* grow = &Gb[w * GBS + gd0];
#pragma unroll
        for (int i = 0; i < 4; ++i) grow[i]     = (__bf16)g0[i];
#pragma unroll
        for (int i = 0; i < 4; ++i) grow[4 + i] = (__bf16)g1[i];
        __syncthreads();

        // prefetch next n (latency hidden behind WMMA phase)
        v4f p0 = g0, p1 = g1;
        if (ni + 1 < NCHUNK) {
            const float* gn = G + ((size_t)(n + 1) * BATCH + (b0 + w)) * DIM + gd0;
            p0 = *(const v4f*)(gn);
            p1 = *(const v4f*)(gn + 4);
        }

        // pre-activation tile: C[16x16] = Gtile(16x256) x wk^T(256x16), K-loop of 8
        v8f c = {};
#pragma unroll
        for (int j = 0; j < 8; ++j) {
            const __bf16* ap = &Gb[l15 * GBS + 32 * j + hi * 8];
            V16 a;
            a.h[0] = *(const v8bf*)(ap);        // K = 32j+koff+0..7
            a.h[1] = *(const v8bf*)(ap + 16);   // K = 32j+koff+16..23
            c = __builtin_amdgcn_wmma_f32_16x16x32_bf16(
                    false, a.v, false, bfrag[j].v, (short)0, c, false, false);
        }

        // epilogue: + (q + wk_b), tanh, dot with wg column weight
        const int m0 = hi * 8;
        float rs[8];
#pragma unroll
        for (int r = 0; r < 8; ++r) {
            float p = c[r] + qs[(m0 + r) * DIM + ncol];
            p = fminf(fmaxf(p, -10.f), 10.f);
            float ex = __expf(2.f * p);
            float y  = (ex - 1.f) * __builtin_amdgcn_rcpf(ex + 1.f);
            rs[r] = y * wgv;
        }
        // reduce over the 16 columns within each half-wave
#pragma unroll
        for (int m = 1; m < 16; m <<= 1) {
#pragma unroll
            for (int r = 0; r < 8; ++r) rs[r] += __shfl_xor(rs[r], m, 32);
        }
        if (l15 == 0) {
#pragma unroll
            for (int r = 0; r < 8; ++r) mpart[w][m0 + r] = rs[r];
        }
        __syncthreads();

        // online softmax stats (one lane per batch row), deterministic order
        if (tid < BT) {
            float msum = 0.f;
#pragma unroll
            for (int ww = 0; ww < 16; ++ww) msum += mpart[ww][tid];
            ws_m[(size_t)(b0 + tid) * NSTEPS + n] = msum;
            float om = runmax[tid];
            float nm = fmaxf(om, msum);
            float wt = __expf(msum - nm);
            float rf = __expf(om - nm);      // exp(-inf)=0 on first step
            runsum[tid] = runsum[tid] * rf + wt;
            runmax[tid] = nm;
            wsh[tid] = wt; rfsh[tid] = rf;
        }
        __syncthreads();

        // z accumulation with online rescale, from fp32 G still in registers
        const float rf = rfsh[w], wt = wsh[w];
        float* zr = &zsh[w * DIM + gd0];
#pragma unroll
        for (int i = 0; i < 4; ++i) zr[i]     = zr[i]     * rf + wt * g0[i];
#pragma unroll
        for (int i = 0; i < 4; ++i) zr[4 + i] = zr[4 + i] * rf + wt * g1[i];

        g0 = p0; g1 = p1;
    }
    __syncthreads();

    for (int i = tid; i < BT * DIM; i += 512) {
        ws_z[((size_t)chunk * BATCH + b0 + (i >> 8)) * DIM + (i & 255)] = zsh[i];
    }
    if (tid < BT) {
        size_t s = ((size_t)chunk * BATCH + b0 + tid) * 2;
        ws_stats[s]     = runmax[tid];
        ws_stats[s + 1] = runsum[tid];
    }
}

// ---------------------------------------------------------------- combine chunks
__global__ __launch_bounds__(256)
void k_combine(const float* __restrict__ ws_m, const float* __restrict__ ws_z,
               const float* __restrict__ ws_stats,
               float* __restrict__ outA, float* __restrict__ outZ) {
    const int b = blockIdx.x;
    const int t = threadIdx.x;
    float mx[NSPLIT], sm[NSPLIT];
    float M = -__builtin_inff();
#pragma unroll
    for (int k = 0; k < NSPLIT; ++k) {
        size_t s = ((size_t)k * BATCH + b) * 2;
        mx[k] = ws_stats[s];
        sm[k] = ws_stats[s + 1];
        M = fmaxf(M, mx[k]);
    }
    float S = 0.f, ek[NSPLIT];
#pragma unroll
    for (int k = 0; k < NSPLIT; ++k) {
        ek[k] = __expf(mx[k] - M);
        S += sm[k] * ek[k];
    }
    const float invS = 1.f / S;

    // alpha[b][n]  (NSTEPS == 256 == blockDim)
    outA[(size_t)b * NSTEPS + t] = __expf(ws_m[(size_t)b * NSTEPS + t] - M) * invS;

    // z[b][d]
    float acc = 0.f;
#pragma unroll
    for (int k = 0; k < NSPLIT; ++k)
        acc += ws_z[((size_t)k * BATCH + b) * DIM + t] * ek[k];
    outZ[(size_t)b * DIM + t] = acc * invS;
}

// ----------------------------------------------------------------
extern "C" void kernel_launch(void* const* d_in, const int* /*in_sizes*/, int /*n_in*/,
                              void* d_out, int /*out_size*/, void* d_ws, size_t /*ws_size*/,
                              hipStream_t stream) {
    const float* G   = (const float*)d_in[0];
    const float* h_t = (const float*)d_in[1];
    const float* wkw = (const float*)d_in[2];
    const float* wkb = (const float*)d_in[3];
    const float* wqw = (const float*)d_in[4];
    const float* wqb = (const float*)d_in[5];
    const float* wgw = (const float*)d_in[6];
    // d_in[7] = wg_b: constant logit shift, cancels in softmax

    float* wsf = (float*)d_ws;
    float*  q    = wsf + WS_Q;
    float*  m    = wsf + WS_M;
    float*  z    = wsf + WS_Z;
    float*  st   = wsf + WS_STATS;
    __bf16* wkbf = (__bf16*)(wsf + WS_WKBF);

    float* outA = (float*)d_out;
    float* outZ = outA + (size_t)BATCH * NSTEPS;

    k_conv_wk<<<(DIM * DIM) / 256, 256, 0, stream>>>(wkw, wkbf);
    k_q<<<BATCH / BT, 256, 0, stream>>>(h_t, wqw, wqb, q);
    k_main<<<(BATCH / BT) * NSPLIT, 512, 0, stream>>>(G, q, wkb, wkbf, wgw, m, z, st);
    k_combine<<<BATCH, 256, 0, stream>>>(m, z, st, outA, outZ);
}
